// SelfAttention_3762391351313
// MI455X (gfx1250) — matrix operand order repaired and checked by reading, not compiled, not run on hardware
//
#include <hip/hip_runtime.h>
#include <math.h>

// Problem constants (match reference)
#define B_ 16
#define C_ 512
#define T_ 2048
#define K_ 64

typedef __attribute__((ext_vector_type(16))) __bf16 v16bf;
typedef __attribute__((ext_vector_type(8)))  __bf16 bf8;
typedef __attribute__((ext_vector_type(4)))  __bf16 bf4;
typedef __attribute__((ext_vector_type(8)))  float  v8f;

union ABPack { v16bf v; bf8 h[2]; };

// ---------------------------------------------------------------------------
// CDNA5 async global->LDS copy (16B per lane), tracked with ASYNCcnt.
// vdst = VGPR holding LDS byte address, vaddr = 64-bit global address.
// ---------------------------------------------------------------------------
__device__ __forceinline__ void async_b128(void* lds, const void* g) {
    asm volatile("global_load_async_to_lds_b128 %0, %1, off"
                 :: "v"((uint32_t)(uintptr_t)lds),
                    "v"((uint64_t)(uintptr_t)g)
                 : "memory");
}

// ---------------------------------------------------------------------------
// Kernel 1: f32 -> bf16 conversion (float4 chunks)
// ---------------------------------------------------------------------------
__global__ __launch_bounds__(256) void cvt_f32_bf16(const float* __restrict__ in,
                                                    __bf16* __restrict__ out, int n4) {
    int i = blockIdx.x * blockDim.x + threadIdx.x;
    if (i < n4) {
        float4 f = ((const float4*)in)[i];
        bf4 o = { (__bf16)f.x, (__bf16)f.y, (__bf16)f.z, (__bf16)f.w };
        ((bf4*)out)[i] = o;
    }
}

// ---------------------------------------------------------------------------
// Kernel 2: projections.  kT/qT[(b,t,d)] = (W @ v)[d,t] + bias[d], transposed
// (T x 64) bf16 so downstream GEMMs get contiguous per-lane operand loads.
// ---------------------------------------------------------------------------
__global__ __launch_bounds__(128) void proj_kernel(
    const float*  __restrict__ v,
    const __bf16* __restrict__ Wkb, const __bf16* __restrict__ Wqb,
    const float*  __restrict__ bk,  const float*  __restrict__ bq,
    __bf16* __restrict__ kT, __bf16* __restrict__ qT)
{
    __shared__ alignas(16) __bf16 tile[16][32];   // [t][c] bf16

    const int b    = blockIdx.x / (T_ / 16);
    const int t0   = (blockIdx.x % (T_ / 16)) * 16;
    const int tid  = threadIdx.x;
    const int lane = tid & 31;
    const int wave = tid >> 5;
    const int lh   = lane & 15;
    const bool hi  = lane >= 16;
    const int mt   = wave;                        // d-tile index 0..3

    v8f accK = {};
    v8f accQ = {};
    const int arow = mt * 16 + lh;                // W row (output channel d)

    for (int c0 = 0; c0 < C_; c0 += 32) {
        {
            const int c_loc = tid >> 2;           // 0..31
            const int t4    = (tid & 3) << 2;     // 0,4,8,12
            const float4 f = *(const float4*)&v[((size_t)b * C_ + c0 + c_loc) * T_ + t0 + t4];
            tile[t4 + 0][c_loc] = (__bf16)f.x;
            tile[t4 + 1][c_loc] = (__bf16)f.y;
            tile[t4 + 2][c_loc] = (__bf16)f.z;
            tile[t4 + 3][c_loc] = (__bf16)f.w;
        }
        __syncthreads();

        ABPack aK, aQ, bV;
        const int ca = c0 + (hi ? 8 : 0);
        aK.h[0] = *(const bf8*)&Wkb[(size_t)arow * C_ + ca];
        aK.h[1] = *(const bf8*)&Wkb[(size_t)arow * C_ + ca + 16];
        aQ.h[0] = *(const bf8*)&Wqb[(size_t)arow * C_ + ca];
        aQ.h[1] = *(const bf8*)&Wqb[(size_t)arow * C_ + ca + 16];

        const int kk0 = hi ? 8 : 0;
        bV.h[0] = *(const bf8*)&tile[lh][kk0];
        bV.h[1] = *(const bf8*)&tile[lh][kk0 + 16];

        accK = __builtin_amdgcn_wmma_f32_16x16x32_bf16(false, aK.v, false, bV.v,
                                                       (short)0, accK, false, false);
        accQ = __builtin_amdgcn_wmma_f32_16x16x32_bf16(false, aQ.v, false, bV.v,
                                                       (short)0, accQ, false, false);
        __syncthreads();
    }

    const int dbase = mt * 16 + (hi ? 8 : 0);
    const int t = t0 + lh;
    bf8 ok, oq;
    #pragma unroll
    for (int r = 0; r < 8; ++r) {
        ok[r] = (__bf16)(accK[r] + bk[dbase + r]);
        oq[r] = (__bf16)(accQ[r] + bq[dbase + r]);
    }
    *(bf8*)&kT[((size_t)b * T_ + t) * K_ + dbase] = ok;
    *(bf8*)&qT[((size_t)b * T_ + t) * K_ + dbase] = oq;
}

// ---------------------------------------------------------------------------
// Kernel 3: scores (one batch).  sT[j,i] = sum_d qT[j,d] * kT[i,d]   (f32)
// ---------------------------------------------------------------------------
__global__ __launch_bounds__(128) void scores_kernel(
    const __bf16* __restrict__ kT, const __bf16* __restrict__ qT,
    float* __restrict__ sT, int b)
{
    const int lane = threadIdx.x & 31;
    const int wave = threadIdx.x >> 5;
    const int lh   = lane & 15;
    const bool hi  = lane >= 16;
    const int j0 = (blockIdx.x % (T_ / 16)) * 16;
    const int i0 = (blockIdx.x / (T_ / 16)) * 64 + wave * 16;

    const __bf16* kb = kT + (size_t)b * T_ * K_;
    const __bf16* qb = qT + (size_t)b * T_ * K_;

    v8f acc = {};
    #pragma unroll
    for (int c0 = 0; c0 < K_; c0 += 32) {
        ABPack a, bm;
        const int ca = c0 + (hi ? 8 : 0);
        a.h[0] = *(const bf8*)&qb[(size_t)(j0 + lh) * K_ + ca];
        a.h[1] = *(const bf8*)&qb[(size_t)(j0 + lh) * K_ + ca + 16];
        const int cb = c0 + (hi ? 8 : 0);
        bm.h[0] = *(const bf8*)&kb[(size_t)(i0 + lh) * K_ + cb];
        bm.h[1] = *(const bf8*)&kb[(size_t)(i0 + lh) * K_ + cb + 16];
        acc = __builtin_amdgcn_wmma_f32_16x16x32_bf16(false, a.v, false, bm.v,
                                                      (short)0, acc, false, false);
    }
    const int jb = j0 + (hi ? 8 : 0);
    const int i  = i0 + lh;
    #pragma unroll
    for (int r = 0; r < 8; ++r)
        sT[(size_t)(jb + r) * T_ + i] = acc[r];
}

// ---------------------------------------------------------------------------
// Kernel 4: softmax over i for each row j of sT (2048 elems); emits bf16 aT.
// ---------------------------------------------------------------------------
__global__ __launch_bounds__(256) void softmax_kernel(const float* __restrict__ sT,
                                                      __bf16* __restrict__ aT)
{
    __shared__ float red[256];
    const int j   = blockIdx.x;
    const int tid = threadIdx.x;
    const float* row = sT + (size_t)j * T_;

    float4 x0 = *(const float4*)&row[tid * 8];
    float4 x1 = *(const float4*)&row[tid * 8 + 4];

    float m = fmaxf(fmaxf(fmaxf(x0.x, x0.y), fmaxf(x0.z, x0.w)),
                    fmaxf(fmaxf(x1.x, x1.y), fmaxf(x1.z, x1.w)));
    red[tid] = m;
    __syncthreads();
    for (int s = 128; s > 0; s >>= 1) {
        if (tid < s) red[tid] = fmaxf(red[tid], red[tid + s]);
        __syncthreads();
    }
    m = red[0];
    __syncthreads();

    float e[8];
    e[0] = __expf(x0.x - m); e[1] = __expf(x0.y - m);
    e[2] = __expf(x0.z - m); e[3] = __expf(x0.w - m);
    e[4] = __expf(x1.x - m); e[5] = __expf(x1.y - m);
    e[6] = __expf(x1.z - m); e[7] = __expf(x1.w - m);

    float sum = (e[0] + e[1]) + (e[2] + e[3]) + (e[4] + e[5]) + (e[6] + e[7]);
    red[tid] = sum;
    __syncthreads();
    for (int s = 128; s > 0; s >>= 1) {
        if (tid < s) red[tid] += red[tid + s];
        __syncthreads();
    }
    const float inv = 1.0f / red[0];

    bf8 o;
    #pragma unroll
    for (int r = 0; r < 8; ++r) o[r] = (__bf16)(e[r] * inv);
    *(bf8*)&aT[(size_t)j * T_ + tid * 8] = o;
}

// ---------------------------------------------------------------------------
// Kernel 5: attention apply (one batch), LDS-tiled + async double buffering.
//   out[c,t] = v[c,t] + gamma * sum_i v_bf[c,i] * aT[t,i]     (inner = 2048)
// Block = 256 threads (8 waves).  Block tile: 128 c x 64 t.
// Wave grid 4(m) x 2(n); each wave owns 2x2 WMMA accumulator tiles.
// A/B tiles double-buffered in LDS, filled via global_load_async_to_lds_b128
// (ASYNCcnt), consumed via ds_load_b128.
// ---------------------------------------------------------------------------
__global__ __launch_bounds__(256) void att_kernel(
    const float*  __restrict__ v,  const __bf16* __restrict__ vb,
    const __bf16* __restrict__ aT, const float*  __restrict__ gamma,
    float* __restrict__ out, int b)
{
    __shared__ alignas(16) __bf16 tileA[2][128][32];   // v_bf  [c][i]  16 KB
    __shared__ alignas(16) __bf16 tileB[2][64][32];    // aT    [t][i]   8 KB

    const int tid  = threadIdx.x;
    const int lane = tid & 31;
    const int wave = tid >> 5;
    const int lh   = lane & 15;
    const bool hi  = lane >= 16;
    const int wm   = wave & 3;                 // 0..3  -> c sub-block of 32
    const int wn   = wave >> 2;                // 0..1  -> t sub-block of 32

    const int t0 = (blockIdx.x % (T_ / 64)) * 64;
    const int c0 = (blockIdx.x / (T_ / 64)) * 128;

    const __bf16* vbb = vb + (size_t)b * C_ * T_;

    // Stage one 128x32 A tile (2 async instrs) + one 64x32 B tile (1 instr).
    auto issue = [&](int s, int i0) {
        #pragma unroll
        for (int k2 = 0; k2 < 2; ++k2) {
            const int chunk = tid + 256 * k2;          // 0..511
            const int row = chunk >> 2, c4 = chunk & 3;
            async_b128(&tileA[s][row][c4 * 8],
                       &vbb[(size_t)(c0 + row) * T_ + i0 + c4 * 8]);
        }
        {
            const int row = tid >> 2, c4 = tid & 3;    // 64 rows x 4 chunks
            async_b128(&tileB[s][row][c4 * 8],
                       &aT[(size_t)(t0 + row) * T_ + i0 + c4 * 8]);
        }
    };

    v8f acc[2][2] = {};

    issue(0, 0);
    const int NSTEP = T_ / 32;                 // 64
    for (int step = 0; step < NSTEP; ++step) {
        const int cur = step & 1;
        const bool last = (step == NSTEP - 1);
        if (!last) issue(cur ^ 1, (step + 1) * 32);

        // This wave's own async copies for stage `cur` are complete when
        // ASYNCcnt <= 3 (<=0 on the last step); barrier makes them visible
        // block-wide before the ds reads.
        if (last) asm volatile("s_wait_asynccnt 0x0" ::: "memory");
        else      asm volatile("s_wait_asynccnt 0x3" ::: "memory");
        __syncthreads();

        ABPack ap[2], bp[2];
        #pragma unroll
        for (int mtt = 0; mtt < 2; ++mtt) {
            const __bf16* ar = &tileA[cur][wm * 32 + mtt * 16 + lh][hi ? 8 : 0];
            ap[mtt].h[0] = *(const bf8*)ar;
            ap[mtt].h[1] = *(const bf8*)(ar + 16);
        }
        #pragma unroll
        for (int ntt = 0; ntt < 2; ++ntt) {
            const __bf16* br = &tileB[cur][wn * 32 + ntt * 16 + lh][hi ? 8 : 0];
            bp[ntt].h[0] = *(const bf8*)br;
            bp[ntt].h[1] = *(const bf8*)(br + 16);
        }
        #pragma unroll
        for (int mtt = 0; mtt < 2; ++mtt)
            #pragma unroll
            for (int ntt = 0; ntt < 2; ++ntt)
                acc[mtt][ntt] = __builtin_amdgcn_wmma_f32_16x16x32_bf16(
                    false, ap[mtt].v, false, bp[ntt].v,
                    (short)0, acc[mtt][ntt], false, false);
        __syncthreads();
    }

    // Epilogue: residual + gamma scale, f32 stores.
    const float g = gamma[0];
    #pragma unroll
    for (int mtt = 0; mtt < 2; ++mtt) {
        const int cb = c0 + wm * 32 + mtt * 16 + (hi ? 8 : 0);
        #pragma unroll
        for (int ntt = 0; ntt < 2; ++ntt) {
            const int t = t0 + wn * 32 + ntt * 16 + lh;
            #pragma unroll
            for (int r = 0; r < 8; ++r) {
                const size_t idx = ((size_t)b * C_ + cb + r) * T_ + t;
                out[idx] = v[idx] + g * acc[mtt][ntt][r];
            }
        }
    }
}

// ---------------------------------------------------------------------------
// Launch: inputs (v, Wk, bk, Wq, bq, gamma); output (B,C,T) f32.
// Workspace: v_bf 32MiB | kT 4MiB | qT 4MiB | sT 16MiB | aT 8MiB | W 128KiB
// ---------------------------------------------------------------------------
extern "C" void kernel_launch(void* const* d_in, const int* in_sizes, int n_in,
                              void* d_out, int out_size, void* d_ws, size_t ws_size,
                              hipStream_t stream) {
    const float* v     = (const float*)d_in[0];
    const float* Wk    = (const float*)d_in[1];
    const float* bk    = (const float*)d_in[2];
    const float* Wq    = (const float*)d_in[3];
    const float* bq    = (const float*)d_in[4];
    const float* gamma = (const float*)d_in[5];
    float* out = (float*)d_out;

    char* ws = (char*)d_ws;
    size_t off = 0;
    __bf16* v_bf = (__bf16*)(ws + off); off += (size_t)B_ * C_ * T_ * 2;   // 32 MiB
    __bf16* kT   = (__bf16*)(ws + off); off += (size_t)B_ * T_ * K_ * 2;   // 4 MiB
    __bf16* qT   = (__bf16*)(ws + off); off += (size_t)B_ * T_ * K_ * 2;   // 4 MiB
    float*  sT   = (float*) (ws + off); off += (size_t)T_ * T_ * 4;        // 16 MiB
    __bf16* aT   = (__bf16*)(ws + off); off += (size_t)T_ * T_ * 2;        // 8 MiB
    __bf16* Wkb  = (__bf16*)(ws + off); off += (size_t)K_ * C_ * 2;
    __bf16* Wqb  = (__bf16*)(ws + off); off += (size_t)K_ * C_ * 2;

    {
        int n4 = (B_ * C_ * T_) / 4;
        cvt_f32_bf16<<<(n4 + 255) / 256, 256, 0, stream>>>(v, v_bf, n4);
        int w4 = (K_ * C_) / 4;
        cvt_f32_bf16<<<(w4 + 255) / 256, 256, 0, stream>>>(Wk, Wkb, w4);
        cvt_f32_bf16<<<(w4 + 255) / 256, 256, 0, stream>>>(Wq, Wqb, w4);
    }

    proj_kernel<<<B_ * (T_ / 16), 128, 0, stream>>>(v, Wkb, Wqb, bk, bq, kT, qT);

    for (int b = 0; b < B_; ++b) {
        scores_kernel<<<(T_ / 16) * (T_ / 64), 128, 0, stream>>>(kT, qT, sT, b);
        softmax_kernel<<<T_, 256, 0, stream>>>(sT, aT);
        att_kernel<<<(T_ / 64) * (C_ / 128), 256, 0, stream>>>(v, v_bf, aT, gamma, out, b);
    }
}
